// CGCNN_17059610100461
// MI455X (gfx1250) — matrix operand order repaired and checked by reading, not compiled, not run on hardware
//
#include <hip/hip_runtime.h>
#include <hip/hip_bf16.h>
#include <math.h>

// ---------------- problem constants ----------------
#define NN      100000
#define EE      1600000
#define ORIG    92
#define NBR     41
#define ATOM    64
#define NCONV   3
#define HFEA    128
#define NGRAPH  256
#define ZPAD    192          // 169 padded to multiple of 32 (K for WMMA)
#define WCOLS   128          // gate(64) || core(64)

typedef _Float16 hv4  __attribute__((ext_vector_type(4)));
typedef _Float16 hv8  __attribute__((ext_vector_type(8)));
typedef _Float16 hv16 __attribute__((ext_vector_type(16)));
typedef float    f32x8 __attribute__((ext_vector_type(8)));

union V16U { hv16 v; hv8 h[2]; };

__device__ __forceinline__ float softplusf(float x) {
    return (x > 20.f) ? x : log1pf(__expf(x));
}
__device__ __forceinline__ float sigmoidf(float x) {
    return 1.0f / (1.0f + __expf(-x));
}

// ---------------- small kernels ----------------
__global__ void zeroKernel(float* p, int n) {
    int i = blockIdx.x * blockDim.x + threadIdx.x;
    if (i < n) p[i] = 0.f;
}

// h = x @ W_emb + b_emb   (one node per 64-thread block)
__global__ __launch_bounds__(64) void embedKernel(
    const float* __restrict__ x, const float* __restrict__ W,
    const float* __restrict__ b, float* __restrict__ h)
{
    const int n = blockIdx.x;
    const int t = threadIdx.x;
    __shared__ float sx[ORIG];
    for (int k = t; k < ORIG; k += 64) sx[k] = x[(size_t)n * ORIG + k];
    __syncthreads();
    float acc = b[t];
    #pragma unroll 4
    for (int k = 0; k < ORIG; ++k) acc = fmaf(sx[k], W[k * ATOM + t], acc);
    h[(size_t)n * ATOM + t] = acc;
}

// per-edge distance
__global__ void distKernel(const float* __restrict__ ea, float* __restrict__ dist, int E) {
    int e = blockIdx.x * blockDim.x + threadIdx.x;
    if (e < E) {
        float a = ea[e * 3 + 0], b = ea[e * 3 + 1], c = ea[e * 3 + 2];
        dist[e] = sqrtf(fmaf(a, a, fmaf(b, b, c * c)));
    }
}

// pack Wf||Ws (zero-padded K: 169->192) to f16, layout [l][k][c]
__global__ void packWKernel(const float* __restrict__ Wf, const float* __restrict__ Ws,
                            _Float16* __restrict__ Wcat)
{
    int i = blockIdx.x * blockDim.x + threadIdx.x;
    if (i >= NCONV * ZPAD * WCOLS) return;
    int l = i / (ZPAD * WCOLS);
    int r = i % (ZPAD * WCOLS);
    int k = r / WCOLS;
    int c = r % WCOLS;
    float v = 0.f;
    if (k < 169) v = (c < ATOM) ? Wf[((size_t)l * 169 + k) * ATOM + c]
                                : Ws[((size_t)l * 169 + k) * ATOM + (c - ATOM)];
    Wcat[i] = (_Float16)v;
}

// ---------------- WMMA edge-message kernel ----------------
// block = 64 threads (2 waves), 32 edges/tile (16 per wave), grid-stride over tiles.
__global__ __launch_bounds__(64) void edgeMsgKernel(
    const float* __restrict__ h, const _Float16* __restrict__ Wcat,
    const float* __restrict__ dist, const int* __restrict__ edge_index,
    const float* __restrict__ bf, const float* __restrict__ bs,
    float* __restrict__ agg, int E, int layer)
{
    __shared__ __attribute__((aligned(16))) _Float16 sZ[32][ZPAD];       // 12 KB
    __shared__ __attribute__((aligned(16))) _Float16 sWt[WCOLS * ZPAD];  // 48 KB [c][k]
    __shared__ float sBf[ATOM], sBs[ATOM];

    const int t       = threadIdx.x;
    const int lane    = t & 31;
    const int w       = t >> 5;        // wave id 0..1
    const int lane16  = lane & 15;
    const int halfsel = lane >> 4;
    const int rowBase = w << 4;

    // stage transposed f16 weights once per block
    const _Float16* Wg = Wcat + (size_t)layer * ZPAD * WCOLS;
    for (int i = t; i < ZPAD * WCOLS; i += 64) {
        int k = i >> 7;           // / 128
        int c = i & 127;
        sWt[c * ZPAD + k] = Wg[i];
    }
    sBf[t] = bf[layer * ATOM + (t & 63)];
    sBs[t] = bs[layer * ATOM + (t & 63)];
    __syncthreads();

    const int numTiles = E >> 5;      // 32 edges per tile
    const int m  = t >> 1;            // staged row 0..31
    const int hs = t & 1;             // half-row selector

    for (int tile = blockIdx.x; tile < numTiles; tile += gridDim.x) {
        const int e0 = tile << 5;

        // ---- stage z tile: [h[dst](64) | h[src](64) | rbf(41) | pad(23)] in f16 ----
        {
            const int e    = e0 + m;
            const int srcN = edge_index[e];
            const int dstN = edge_index[E + e];
            const float4* dr = (const float4*)(h + (size_t)dstN * ATOM);
            const float4* sr = (const float4*)(h + (size_t)srcN * ATOM);
            hv4* zr = (hv4*)(&sZ[m][0]);
            if (hs == 0) {
                #pragma unroll
                for (int q = 0; q < 16; ++q) {
                    float4 f = dr[q];
                    zr[q] = (hv4){(_Float16)f.x, (_Float16)f.y, (_Float16)f.z, (_Float16)f.w};
                }
                #pragma unroll
                for (int q = 0; q < 8; ++q) {
                    float4 f = sr[q];
                    zr[16 + q] = (hv4){(_Float16)f.x, (_Float16)f.y, (_Float16)f.z, (_Float16)f.w};
                }
            } else {
                #pragma unroll
                for (int q = 8; q < 16; ++q) {
                    float4 f = sr[q];
                    zr[16 + q] = (hv4){(_Float16)f.x, (_Float16)f.y, (_Float16)f.z, (_Float16)f.w};
                }
                const float d = dist[e];
                #pragma unroll 8
                for (int k = 0; k < NBR; ++k) {         // exp(-5*(d - 0.2k)^2)
                    float u = d - 0.2f * (float)k;
                    sZ[m][2 * ATOM + k] = (_Float16)__expf(-5.0f * u * u);
                }
                for (int k = 2 * ATOM + NBR; k < ZPAD; ++k) sZ[m][k] = (_Float16)0.f;
            }
        }
        __syncthreads();

        // ---- WMMA: [16 x 192] x [192 x 128] per wave ----
        f32x8 acc[8];
        #pragma unroll
        for (int n = 0; n < 8; ++n)
            acc[n] = (f32x8){0.f, 0.f, 0.f, 0.f, 0.f, 0.f, 0.f, 0.f};

        const int row = rowBase + lane16;
        #pragma unroll
        for (int kk = 0; kk < 6; ++kk) {
            const int K0 = kk * 32;
            V16U a;   // A layout: lane holds row=lane%16; 16 contiguous K at half*16
            a.h[0] = *(const hv8*)&sZ[row][K0 + halfsel * 16];
            a.h[1] = *(const hv8*)&sZ[row][K0 + halfsel * 16 + 8];
            #pragma unroll
            for (int n = 0; n < 8; ++n) {
                V16U b;  // B layout: lane holds col=lane%16; 16 contiguous K at half*16
                const _Float16* wc = &sWt[(n * 16 + lane16) * ZPAD + K0 + halfsel * 16];
                b.h[0] = *(const hv8*)(wc);
                b.h[1] = *(const hv8*)(wc + 8);
                acc[n] = __builtin_amdgcn_wmma_f32_16x16x32_f16(
                    false, a.v, false, b.v, (short)0, acc[n], false, false);
            }
        }

        // ---- epilogue: gate(col)=acc[n], core(col)=acc[n+4] live in the SAME lane ----
        int dstIds[8];
        #pragma unroll
        for (int r = 0; r < 8; ++r)
            dstIds[r] = edge_index[E + e0 + rowBase + r + 8 * halfsel];
        #pragma unroll
        for (int n = 0; n < 4; ++n) {
            const int col = n * 16 + lane16;
            const float gb = sBf[col];
            const float sb = sBs[col];
            #pragma unroll
            for (int r = 0; r < 8; ++r) {
                float gate = sigmoidf(acc[n][r] + gb);
                float core = softplusf(acc[n + 4][r] + sb);
                atomicAdd(&agg[(size_t)dstIds[r] * ATOM + col], gate * core);
            }
        }
        __syncthreads();
    }
}

// ---------------- BatchNorm stats / finalize / update ----------------
__global__ __launch_bounds__(256) void statsKernel(const float* __restrict__ agg,
                                                   float* __restrict__ stats, int N)
{
    const int c  = threadIdx.x & 63;
    const int rg = threadIdx.x >> 6;   // 0..3
    float s = 0.f, s2 = 0.f;
    for (int n = blockIdx.x * 4 + rg; n < N; n += gridDim.x * 4) {
        float v = agg[(size_t)n * ATOM + c];
        s += v; s2 += v * v;
    }
    __shared__ float red[2][4][64];
    red[0][rg][c] = s; red[1][rg][c] = s2;
    __syncthreads();
    if (rg == 0) {
        s  = red[0][0][c] + red[0][1][c] + red[0][2][c] + red[0][3][c];
        s2 = red[1][0][c] + red[1][1][c] + red[1][2][c] + red[1][3][c];
        atomicAdd(&stats[c], s);
        atomicAdd(&stats[64 + c], s2);
    }
}

__global__ void finalizeStatsKernel(float* stats, int N) {
    int c = threadIdx.x;
    if (c < 64) {
        float inv = 1.0f / (float)N;
        float mu  = stats[c] * inv;
        float var = stats[64 + c] * inv - mu * mu;
        stats[128 + c] = mu;
        stats[192 + c] = rsqrtf(var + 1e-5f);
    }
}

// h = softplus(gamma*(agg-mu)*rstd + beta + h)
__global__ void bnResidualKernel(const float* __restrict__ agg, float* __restrict__ h,
                                 const float* __restrict__ stats,
                                 const float* __restrict__ gamma, const float* __restrict__ beta,
                                 int layer, int count)
{
    int i = blockIdx.x * blockDim.x + threadIdx.x;
    if (i >= count) return;
    int c = i & 63;
    float mu   = stats[128 + c];
    float rstd = stats[192 + c];
    float v = fmaf(gamma[layer * ATOM + c], (agg[i] - mu) * rstd,
                   beta[layer * ATOM + c]) + h[i];
    h[i] = softplusf(v);
}

// global_add_pool
__global__ void poolKernel(const float* __restrict__ h, const int* __restrict__ batch,
                           float* __restrict__ g, int count)
{
    int i = blockIdx.x * blockDim.x + threadIdx.x;
    if (i >= count) return;
    int n = i >> 6, c = i & 63;
    atomicAdd(&g[(size_t)batch[n] * ATOM + c], h[i]);
}

// out = softplus(g@W1 + b1) @ W2 + b2 ; one block per graph
__global__ __launch_bounds__(128) void headKernel(
    const float* __restrict__ g, const float* __restrict__ W1, const float* __restrict__ b1,
    const float* __restrict__ W2, const float* __restrict__ b2, float* __restrict__ out)
{
    const int gr = blockIdx.x;
    const int j  = threadIdx.x;
    __shared__ float sg[ATOM];
    __shared__ float red[HFEA];
    if (j < ATOM) sg[j] = g[(size_t)gr * ATOM + j];
    __syncthreads();
    float acc = b1[j];
    #pragma unroll 4
    for (int c = 0; c < ATOM; ++c) acc = fmaf(sg[c], W1[c * HFEA + j], acc);
    red[j] = softplusf(acc) * W2[j];
    __syncthreads();
    for (int s = HFEA / 2; s > 0; s >>= 1) {
        if (j < s) red[j] += red[j + s];
        __syncthreads();
    }
    if (j == 0) out[gr] = red[0] + b2[0];
}

// ---------------- host ----------------
extern "C" void kernel_launch(void* const* d_in, const int* in_sizes, int n_in,
                              void* d_out, int out_size, void* d_ws, size_t ws_size,
                              hipStream_t stream)
{
    const float* x        = (const float*)d_in[0];
    const float* edgeattr = (const float*)d_in[1];
    const float* W_emb    = (const float*)d_in[2];
    const float* b_emb    = (const float*)d_in[3];
    const float* Wf       = (const float*)d_in[4];
    const float* bf       = (const float*)d_in[5];
    const float* Ws       = (const float*)d_in[6];
    const float* bs       = (const float*)d_in[7];
    const float* bn_gamma = (const float*)d_in[8];
    const float* bn_beta  = (const float*)d_in[9];
    const float* W1       = (const float*)d_in[10];
    const float* b1       = (const float*)d_in[11];
    const float* W2       = (const float*)d_in[12];
    const float* b2       = (const float*)d_in[13];
    const int*   edge_index = (const int*)d_in[14];
    const int*   batch      = (const int*)d_in[15];
    float* out = (float*)d_out;

    char* wsp = (char*)d_ws;
    size_t off = 0;
    auto walloc = [&](size_t bytes) -> void* {
        void* p = wsp + off;
        off = (off + bytes + 255) & ~(size_t)255;
        return p;
    };
    float*     h     = (float*)walloc((size_t)NN * ATOM * 4);
    float*     agg   = (float*)walloc((size_t)NN * ATOM * 4);
    float*     dist  = (float*)walloc((size_t)EE * 4);
    float*     stats = (float*)walloc(256 * 4);
    float*     g     = (float*)walloc((size_t)NGRAPH * ATOM * 4);
    _Float16*  Wcat  = (_Float16*)walloc((size_t)NCONV * ZPAD * WCOLS * 2);

    // embedding + distances + weight pack
    embedKernel<<<NN, 64, 0, stream>>>(x, W_emb, b_emb, h);
    distKernel<<<(EE + 255) / 256, 256, 0, stream>>>(edgeattr, dist, EE);
    packWKernel<<<(NCONV * ZPAD * WCOLS + 255) / 256, 256, 0, stream>>>(Wf, Ws, Wcat);

    const int NC = NN * ATOM;
    for (int l = 0; l < NCONV; ++l) {
        zeroKernel<<<(NC + 255) / 256, 256, 0, stream>>>(agg, NC);
        zeroKernel<<<1, 256, 0, stream>>>(stats, 256);
        edgeMsgKernel<<<4096, 64, 0, stream>>>(h, Wcat, dist, edge_index, bf, bs, agg, EE, l);
        statsKernel<<<1024, 256, 0, stream>>>(agg, stats, NN);
        finalizeStatsKernel<<<1, 64, 0, stream>>>(stats, NN);
        bnResidualKernel<<<(NC + 255) / 256, 256, 0, stream>>>(agg, h, stats, bn_gamma, bn_beta, l, NC);
    }

    zeroKernel<<<(NGRAPH * ATOM + 255) / 256, 256, 0, stream>>>(g, NGRAPH * ATOM);
    poolKernel<<<(NC + 255) / 256, 256, 0, stream>>>(h, batch, g, NC);
    headKernel<<<NGRAPH, HFEA, 0, stream>>>(g, W1, b1, W2, b2, out);
}
